// GNN_VCG_15839839387886
// MI455X (gfx1250) — matrix operand order repaired and checked by reading, not compiled, not run on hardware
//
#include <hip/hip_runtime.h>
#include <hip/hip_bf16.h>

// ---------------------------------------------------------------------------
// NeuroSAT-style GNN forward for MI455X (gfx1250, wave32, WMMA).
// All matmuls run through v_wmma_f32_16x16x32_f16. Weights pre-packed to f16
// B-fragment layout; activations staged into LDS as packed f16 A-fragments.
// Row dimensions are padded (V -> 30016) so all WMMA loops are branchless.
// ---------------------------------------------------------------------------

#define NV_  30000          // real variable count
#define NVP_ 30016          // padded to multiple of 64
#define NC_  120000         // already multiple of 64
#define NEP_ 180000
#define NEN_ 180000
#define DD   128
#define NB_  32
#define NITER 3

typedef __attribute__((ext_vector_type(16))) _Float16 v16h;
typedef __attribute__((ext_vector_type(8)))  _Float16 v8h;
typedef __attribute__((ext_vector_type(8)))  float    v8f;

__device__ __forceinline__ v8f wmma16(v16h a, v16h b, v8f c) {
  return __builtin_amdgcn_wmma_f32_16x16x32_f16(
      false, a, false, b, (short)0, c, false, false);
}

__device__ __forceinline__ float sigmoid_(float x) {
  return 1.0f / (1.0f + __expf(-x));
}

// Packed B fragment: 512 f16 per (ktile,ntile) fragment, 16 contiguous per lane.
__device__ __forceinline__ v16h load_b_packed(const _Float16* __restrict__ P,
                                              int frag, int lane) {
  return *(const v16h*)(P + (size_t)frag * 512 + lane * 16);
}

// Convert 8 consecutive f32 (one "octet" of a row) to f16 at the A-fragment
// position for (row, K=oct*8..oct*8+7).
//   oo = oct & 3 within the 32-wide k-tile:
//   oo=0 -> half 0, ii 0..7 ; oo=1 -> half 1, ii 0..7
//   oo=2 -> half 0, ii 8..15; oo=3 -> half 1, ii 8..15
__device__ __forceinline__ void pack_octet(_Float16* __restrict__ dst8,
                                           float4 x0, float4 x1, float s) {
  v8h h;
  h[0] = (_Float16)(x0.x * s); h[1] = (_Float16)(x0.y * s);
  h[2] = (_Float16)(x0.z * s); h[3] = (_Float16)(x0.w * s);
  h[4] = (_Float16)(x1.x * s); h[5] = (_Float16)(x1.y * s);
  h[6] = (_Float16)(x1.z * s); h[7] = (_Float16)(x1.w * s);
  *(v8h*)dst8 = h;
}

// ---------------------------------------------------------------------------
// Weight pre-pack: row-major f32 (K x N) -> f16 fragments [kt][nt][lane][16].
// ---------------------------------------------------------------------------
__global__ void pack_w_kernel(const float* __restrict__ W, _Float16* __restrict__ P,
                              int N, int nfrag) {
  const int lane = threadIdx.x & 31;
  const int frag = blockIdx.x * (blockDim.x >> 5) + (threadIdx.x >> 5);
  if (frag >= nfrag) return;
  const int ntiles = N >> 4;
  const int kt = frag / ntiles, nt = frag % ntiles;
  const int n = nt * 16 + (lane & 15);
  const int kbase = kt * 32 + (lane >> 4) * 8;
  _Float16* out = P + (size_t)frag * 512 + lane * 16;
#pragma unroll
  for (int i = 0; i < 16; ++i)
    out[i] = (_Float16)W[(size_t)(kbase + i + ((i >> 3) << 3)) * N + n];
}

// ---------------------------------------------------------------------------
// Fused 2-layer MLP: Out = relu(X@W1+b1)@W2+b2, D=128. Block = 64 rows
// (row count padded -> branchless). 8 waves x (16-col tile, 4 row-tiles).
// ---------------------------------------------------------------------------
__global__ void __launch_bounds__(256) mlp_fused_kernel(
    const float* __restrict__ X,
    const _Float16* __restrict__ W1P, const float* __restrict__ b1,
    const _Float16* __restrict__ W2P, const float* __restrict__ b2,
    float* __restrict__ Out) {
  __shared__ __align__(32) _Float16 aP[4][4][32][16];    // 16 KB
  __shared__ __align__(32) _Float16 hidP[4][4][32][16];  // 16 KB

  const int lane = threadIdx.x & 31;
  const int wave = threadIdx.x >> 5;     // 0..7 -> 16-col tile
  const int r0   = blockIdx.x * 64;
  const int m    = lane & 15;
  const int rbase = (lane >> 4) * 8;

  // ---- stage X[r0:r0+64, 0:128] into packed A fragments ----
#pragma unroll
  for (int o = 0; o < 4; ++o) {
    const int idx = threadIdx.x + o * 256;   // 0..1023
    const int row = idx >> 4;                // 0..63
    const int oct = idx & 15;                // K octet
    const int kt = oct >> 2, oo = oct & 3;
    const float* src = X + (size_t)(r0 + row) * DD + oct * 8;
    const float4 x0 = *(const float4*)src;
    const float4 x1 = *(const float4*)(src + 4);
    pack_octet(&aP[row >> 4][kt][(row & 15) + 16 * (oo & 1)][(oo >> 1) * 8],
               x0, x1, 1.0f);
  }
  __syncthreads();

  // ---- stage 1: hidden = relu(X@W1 + b1), packed into hidP ----
  v8f acc[4] = {{0}, {0}, {0}, {0}};
#pragma unroll
  for (int kt = 0; kt < 4; ++kt) {
    const v16h b = load_b_packed(W1P, kt * 8 + wave, lane);
#pragma unroll
    for (int rt = 0; rt < 4; ++rt) {
      const v16h a = *(const v16h*)&aP[rt][kt][lane][0];
      acc[rt] = wmma16(a, b, acc[rt]);
    }
  }
  {
    const int colg = wave * 16 + m;          // hidden column (stage-2 K)
    const int kt = colg >> 5;
    const int k32 = colg & 31;
    const int halfd = (k32 >> 3) & 1;
    const int base = k32 - 8 * halfd;
    const int ii = (base >= 16) ? base - 8 : base;
    const float bv1 = b1[colg];
#pragma unroll
    for (int rt = 0; rt < 4; ++rt)
#pragma unroll
      for (int r = 0; r < 8; ++r)
        hidP[rt][kt][(rbase + r) + 16 * halfd][ii] =
            (_Float16)fmaxf(acc[rt][r] + bv1, 0.0f);
  }
  __syncthreads();

  // ---- stage 2: Out = hidden@W2 + b2 (A from LDS, zero cvt) ----
  v8f acc2[4] = {{0}, {0}, {0}, {0}};
#pragma unroll
  for (int kt = 0; kt < 4; ++kt) {
    const v16h b = load_b_packed(W2P, kt * 8 + wave, lane);
#pragma unroll
    for (int rt = 0; rt < 4; ++rt) {
      const v16h a = *(const v16h*)&hidP[rt][kt][lane][0];
      acc2[rt] = wmma16(a, b, acc2[rt]);
    }
  }
  const int col = wave * 16 + m;
  const float bv2 = b2[col];
#pragma unroll
  for (int rt = 0; rt < 4; ++rt)
#pragma unroll
    for (int r = 0; r < 8; ++r)
      Out[(size_t)(r0 + rt * 16 + rbase + r) * DD + col] = acc2[rt][r] + bv2;
}

// ---------------------------------------------------------------------------
// Fused GRU cell. Block = 32 rows x 4 col-tiles (4 waves), rows padded ->
// branchless. x = normalized [p_aggr, n_aggr] staged packed-f16 in LDS
// (scale folded in); h staged too. 12 accumulators, 72 WMMAs per wave.
// ---------------------------------------------------------------------------
__global__ void __launch_bounds__(128) gru_fused_kernel(
    const float* __restrict__ Aggr,   // N x 2D (unnormalized sums)
    const float* __restrict__ pinv, const float* __restrict__ ninv,
    const float* __restrict__ H,      // N x D
    const _Float16* __restrict__ WihP,  // packed (2D x 3D)
    const _Float16* __restrict__ WhhP,  // packed (D x 3D)
    const float* __restrict__ bih, const float* __restrict__ bhh,
    float* __restrict__ Hout) {
  __shared__ __align__(32) _Float16 aPx[2][8][32][16];  // 16 KB (x, K=256)
  __shared__ __align__(32) _Float16 aPh[2][4][32][16];  //  8 KB (h, K=128)

  const int lane = threadIdx.x & 31;
  const int wave = threadIdx.x >> 5;              // 0..3
  const int pt = blockIdx.x >> 1;                 // 32-row tile
  const int j  = (blockIdx.x & 1) * 4 + wave;     // 16-col tile of D
  const int row0 = pt * 32;
  const int m = lane & 15;

  // ---- stage x fragments (normalization folded in) ----
#pragma unroll
  for (int o = 0; o < 8; ++o) {
    const int idx = threadIdx.x + o * 128;   // 0..1023
    const int row = idx >> 5;                // 0..31
    const int oct = idx & 31;                // K octet (K = oct*8)
    const int grow = row0 + row;
    const float s = (oct < 16) ? pinv[grow] : ninv[grow];
    const int kt = oct >> 2, oo = oct & 3;
    const float* src = Aggr + (size_t)grow * (2 * DD) + oct * 8;
    const float4 x0 = *(const float4*)src;
    const float4 x1 = *(const float4*)(src + 4);
    pack_octet(&aPx[row >> 4][kt][(row & 15) + 16 * (oo & 1)][(oo >> 1) * 8],
               x0, x1, s);
  }
  // ---- stage h fragments ----
#pragma unroll
  for (int o = 0; o < 4; ++o) {
    const int idx = threadIdx.x + o * 128;   // 0..511
    const int row = idx >> 4;                // 0..31
    const int oct = idx & 15;
    const int kt = oct >> 2, oo = oct & 3;
    const float* src = H + (size_t)(row0 + row) * DD + oct * 8;
    const float4 x0 = *(const float4*)src;
    const float4 x1 = *(const float4*)(src + 4);
    pack_octet(&aPh[row >> 4][kt][(row & 15) + 16 * (oo & 1)][(oo >> 1) * 8],
               x0, x1, 1.0f);
  }
  __syncthreads();

  v8f gi_r0 = {0}, gi_z0 = {0}, gi_n0 = {0};
  v8f gi_r1 = {0}, gi_z1 = {0}, gi_n1 = {0};
  v8f gh_r0 = {0}, gh_z0 = {0}, gh_n0 = {0};
  v8f gh_r1 = {0}, gh_z1 = {0}, gh_n1 = {0};

#pragma unroll
  for (int kt = 0; kt < 8; ++kt) {           // gi: K = 256
    const v16h br = load_b_packed(WihP, kt * 24 + j,      lane);
    const v16h bz = load_b_packed(WihP, kt * 24 + j + 8,  lane);
    const v16h bn = load_b_packed(WihP, kt * 24 + j + 16, lane);
    const v16h a0 = *(const v16h*)&aPx[0][kt][lane][0];
    const v16h a1 = *(const v16h*)&aPx[1][kt][lane][0];
    gi_r0 = wmma16(a0, br, gi_r0);
    gi_z0 = wmma16(a0, bz, gi_z0);
    gi_n0 = wmma16(a0, bn, gi_n0);
    gi_r1 = wmma16(a1, br, gi_r1);
    gi_z1 = wmma16(a1, bz, gi_z1);
    gi_n1 = wmma16(a1, bn, gi_n1);
  }
#pragma unroll
  for (int kt = 0; kt < 4; ++kt) {           // gh: K = 128
    const v16h br = load_b_packed(WhhP, kt * 24 + j,      lane);
    const v16h bz = load_b_packed(WhhP, kt * 24 + j + 8,  lane);
    const v16h bn = load_b_packed(WhhP, kt * 24 + j + 16, lane);
    const v16h a0 = *(const v16h*)&aPh[0][kt][lane][0];
    const v16h a1 = *(const v16h*)&aPh[1][kt][lane][0];
    gh_r0 = wmma16(a0, br, gh_r0);
    gh_z0 = wmma16(a0, bz, gh_z0);
    gh_n0 = wmma16(a0, bn, gh_n0);
    gh_r1 = wmma16(a1, br, gh_r1);
    gh_z1 = wmma16(a1, bz, gh_z1);
    gh_n1 = wmma16(a1, bn, gh_n1);
  }

  const int col = j * 16 + m;
  const float br_  = bih[col];            const float bhr = bhh[col];
  const float bz_  = bih[col + DD];       const float bhz = bhh[col + DD];
  const float bn_  = bih[col + 2 * DD];   const float bhn = bhh[col + 2 * DD];
  const int rbase = (lane >> 4) * 8;
#pragma unroll
  for (int r = 0; r < 8; ++r) {
    const int row = row0 + rbase + r;
    const float hv = H[(size_t)row * DD + col];
    const float rg = sigmoid_(gi_r0[r] + br_ + gh_r0[r] + bhr);
    const float zg = sigmoid_(gi_z0[r] + bz_ + gh_z0[r] + bhz);
    const float ng = tanhf(gi_n0[r] + bn_ + rg * (gh_n0[r] + bhn));
    Hout[(size_t)row * DD + col] = (1.0f - zg) * ng + zg * hv;
  }
#pragma unroll
  for (int r = 0; r < 8; ++r) {
    const int row = row0 + 16 + rbase + r;
    const float hv = H[(size_t)row * DD + col];
    const float rg = sigmoid_(gi_r1[r] + br_ + gh_r1[r] + bhr);
    const float zg = sigmoid_(gi_z1[r] + bz_ + gh_z1[r] + bhz);
    const float ng = tanhf(gi_n1[r] + bn_ + rg * (gh_n1[r] + bhn));
    Hout[(size_t)row * DD + col] = (1.0f - zg) * ng + zg * hv;
  }
}

// ---------------------------------------------------------------------------
// Edge / scatter / setup kernels
// ---------------------------------------------------------------------------
__global__ void gather_endpoints_kernel(const int* __restrict__ vi,
                                        const int* __restrict__ ci,
                                        const int* __restrict__ eidx,
                                        int* __restrict__ outv,
                                        int* __restrict__ outc, int n) {
  int t = blockIdx.x * blockDim.x + threadIdx.x;
  if (t >= n) return;
  int e = eidx[t];
  outv[t] = vi[e];
  outc[t] = ci[e];
}

__global__ void count_kernel(const int* __restrict__ idx, float* __restrict__ cnt, int n) {
  int t = blockIdx.x * blockDim.x + threadIdx.x;
  if (t < n) atomicAdd(&cnt[idx[t]], 1.0f);
}

__global__ void invert_kernel(float* __restrict__ c, int n) {
  int t = blockIdx.x * blockDim.x + threadIdx.x;
  if (t < n) c[t] = 1.0f / fmaxf(c[t], 1.0f);
}

__global__ void init_emb_kernel(const float* __restrict__ init,
                                float* __restrict__ emb, int nelem) {
  int t = blockIdx.x * blockDim.x + threadIdx.x;
  if (t >= nelem) return;
  emb[t] = init[t & (DD - 1)] * 0.08838834764831845f;  // 1/sqrt(128)
}

__global__ void scatter_kernel(const float* __restrict__ Msg,
                               const int* __restrict__ src,
                               const int* __restrict__ dst,
                               float* __restrict__ Aggr, int coloff, int nedges) {
  int t = blockIdx.x * blockDim.x + threadIdx.x;
  int e = t >> 5, q = t & 31;
  if (e >= nedges) return;
  int s = src[e], d = dst[e];
  const float4 mm = *(const float4*)(Msg + (size_t)s * DD + q * 4);
  float* o = Aggr + (size_t)d * (2 * DD) + coloff + q * 4;
  atomicAdd(o + 0, mm.x); atomicAdd(o + 1, mm.y);
  atomicAdd(o + 2, mm.z); atomicAdd(o + 3, mm.w);
}

__global__ void readout_sum_kernel(const float* __restrict__ vemb,
                                   const int* __restrict__ vb,
                                   float* __restrict__ gsum,
                                   float* __restrict__ gcnt, int n) {
  int t = blockIdx.x * blockDim.x + threadIdx.x;
  int v = t >> 5, q = t & 31;
  if (v >= n) return;
  int b = vb[v];
  const float4 mm = *(const float4*)(vemb + (size_t)v * DD + q * 4);
  float* o = gsum + (size_t)b * DD + q * 4;
  atomicAdd(o + 0, mm.x); atomicAdd(o + 1, mm.y);
  atomicAdd(o + 2, mm.z); atomicAdd(o + 3, mm.w);
  if (q == 0) atomicAdd(&gcnt[b], 1.0f);
}

__global__ void __launch_bounds__(128) readout_mlp_kernel(
    const float* __restrict__ gsum, const float* __restrict__ gcnt,
    const float* __restrict__ W1, const float* __restrict__ b1,
    const float* __restrict__ W2, const float* __restrict__ b2,
    float* __restrict__ out) {
  __shared__ float red[DD];
  const int d = threadIdx.x;
  for (int g = 0; g < NB_; ++g) {
    const float inv = 1.0f / fmaxf(gcnt[g], 1.0f);
    float acc = b1[d];
    for (int k = 0; k < DD; ++k)
      acc += gsum[(size_t)g * DD + k] * inv * W1[(size_t)k * DD + d];
    red[d] = fmaxf(acc, 0.0f) * W2[d];
    __syncthreads();
    for (int s = 64; s > 0; s >>= 1) {
      if (d < s) red[d] += red[d + s];
      __syncthreads();
    }
    if (d == 0) out[g] = sigmoid_(red[0] + b2[0]);
    __syncthreads();
  }
}

// ---------------------------------------------------------------------------
// Host driver (graph-capture safe: kernel launches + hipMemsetAsync only)
// ---------------------------------------------------------------------------
extern "C" void kernel_launch(void* const* d_in, const int* in_sizes, int n_in,
                              void* d_out, int out_size, void* d_ws, size_t ws_size,
                              hipStream_t stream) {
  (void)in_sizes; (void)n_in; (void)out_size; (void)ws_size;

  const int*   v_edge  = (const int*)d_in[0];
  const int*   c_edge  = (const int*)d_in[1];
  const int*   p_idx   = (const int*)d_in[2];
  const int*   n_idx   = (const int*)d_in[3];
  const int*   v_batch = (const int*)d_in[4];
  const float* v_init  = (const float*)d_in[5];
  const float* c_init  = (const float*)d_in[6];
  const float* mlpW[4][4];
  for (int i = 0; i < 4; ++i)
    for (int jj = 0; jj < 4; ++jj)
      mlpW[i][jj] = (const float*)d_in[7 + i * 4 + jj];
  const float* gc_Wih = (const float*)d_in[23];
  const float* gc_Whh = (const float*)d_in[24];
  const float* gc_bih = (const float*)d_in[25];
  const float* gc_bhh = (const float*)d_in[26];
  const float* gv_Wih = (const float*)d_in[27];
  const float* gv_Whh = (const float*)d_in[28];
  const float* gv_bih = (const float*)d_in[29];
  const float* gv_bhh = (const float*)d_in[30];
  const float* ro_W1  = (const float*)d_in[31];
  const float* ro_b1  = (const float*)d_in[32];
  const float* ro_W2  = (const float*)d_in[33];
  const float* ro_b2  = (const float*)d_in[34];

  // ---- workspace layout (~436 MB), padded row counts ----
  char* w = (char*)d_ws;
  size_t off = 0;
  auto take = [&](size_t bytes) -> void* {
    off = (off + 255) & ~(size_t)255;
    void* p = w + off;
    off += bytes;
    return p;
  };
  float* cembA  = (float*)take((size_t)NC_  * DD * 4);
  float* cembB  = (float*)take((size_t)NC_  * DD * 4);
  float* vembA  = (float*)take((size_t)NVP_ * DD * 4);
  float* vembB  = (float*)take((size_t)NVP_ * DD * 4);
  float* msg_p  = (float*)take((size_t)NC_  * DD * 4);   // sized for max(V,C)
  float* msg_n  = (float*)take((size_t)NC_  * DD * 4);
  float* aggr_c = (float*)take((size_t)NC_  * 2 * DD * 4);
  float* aggr_v = (float*)take((size_t)NVP_ * 2 * DD * 4);
  int*   pv     = (int*)take((size_t)NEP_ * 4);
  int*   pc     = (int*)take((size_t)NEP_ * 4);
  int*   nv     = (int*)take((size_t)NEN_ * 4);
  int*   nc     = (int*)take((size_t)NEN_ * 4);
  float* pcinv  = (float*)take((size_t)NC_  * 4);
  float* ncinv  = (float*)take((size_t)NC_  * 4);
  float* pvinv  = (float*)take((size_t)NVP_ * 4);
  float* nvinv  = (float*)take((size_t)NVP_ * 4);
  float* gsum   = (float*)take((size_t)NB_ * DD * 4);
  float* gcnt   = (float*)take((size_t)NB_ * 4);
  _Float16* mlpP[4][2];
  for (int i = 0; i < 4; ++i) {
    mlpP[i][0] = (_Float16*)take((size_t)32 * 512 * 2);   // W1: 4x8 frags
    mlpP[i][1] = (_Float16*)take((size_t)32 * 512 * 2);   // W2
  }
  _Float16* gcWihP = (_Float16*)take((size_t)192 * 512 * 2);  // 8x24 frags
  _Float16* gcWhhP = (_Float16*)take((size_t)96  * 512 * 2);  // 4x24 frags
  _Float16* gvWihP = (_Float16*)take((size_t)192 * 512 * 2);
  _Float16* gvWhhP = (_Float16*)take((size_t)96  * 512 * 2);

  // ---- pack weights (tiny) ----
  for (int i = 0; i < 4; ++i) {
    pack_w_kernel<<<8, 128, 0, stream>>>(mlpW[i][0], mlpP[i][0], DD, 32);
    pack_w_kernel<<<8, 128, 0, stream>>>(mlpW[i][2], mlpP[i][1], DD, 32);
  }
  pack_w_kernel<<<48, 128, 0, stream>>>(gc_Wih, gcWihP, 3 * DD, 192);
  pack_w_kernel<<<24, 128, 0, stream>>>(gc_Whh, gcWhhP, 3 * DD, 96);
  pack_w_kernel<<<48, 128, 0, stream>>>(gv_Wih, gvWihP, 3 * DD, 192);
  pack_w_kernel<<<24, 128, 0, stream>>>(gv_Whh, gvWhhP, 3 * DD, 96);

  // ---- setup: edge endpoints, degrees, initial embeddings ----
  gather_endpoints_kernel<<<(NEP_ + 255) / 256, 256, 0, stream>>>(v_edge, c_edge, p_idx, pv, pc, NEP_);
  gather_endpoints_kernel<<<(NEN_ + 255) / 256, 256, 0, stream>>>(v_edge, c_edge, n_idx, nv, nc, NEN_);
  hipMemsetAsync(pcinv, 0, (size_t)NC_  * 4, stream);
  hipMemsetAsync(ncinv, 0, (size_t)NC_  * 4, stream);
  hipMemsetAsync(pvinv, 0, (size_t)NVP_ * 4, stream);
  hipMemsetAsync(nvinv, 0, (size_t)NVP_ * 4, stream);
  count_kernel<<<(NEP_ + 255) / 256, 256, 0, stream>>>(pc, pcinv, NEP_);
  count_kernel<<<(NEN_ + 255) / 256, 256, 0, stream>>>(nc, ncinv, NEN_);
  count_kernel<<<(NEP_ + 255) / 256, 256, 0, stream>>>(pv, pvinv, NEP_);
  count_kernel<<<(NEN_ + 255) / 256, 256, 0, stream>>>(nv, nvinv, NEN_);
  invert_kernel<<<(NC_  + 255) / 256, 256, 0, stream>>>(pcinv, NC_);
  invert_kernel<<<(NC_  + 255) / 256, 256, 0, stream>>>(ncinv, NC_);
  invert_kernel<<<(NVP_ + 255) / 256, 256, 0, stream>>>(pvinv, NVP_);
  invert_kernel<<<(NVP_ + 255) / 256, 256, 0, stream>>>(nvinv, NVP_);
  init_emb_kernel<<<(NVP_ * DD + 255) / 256, 256, 0, stream>>>(v_init, vembA, NVP_ * DD);
  init_emb_kernel<<<(NC_  * DD + 255) / 256, 256, 0, stream>>>(c_init, cembA, NC_ * DD);

  float* vcur = vembA; float* vnext = vembB;
  float* ccur = cembA; float* cnext = cembB;

  const int scatterP = (NEP_ * 32 + 255) / 256;
  const int scatterN = (NEN_ * 32 + 255) / 256;
  const int mlpBlocksV = NVP_ / 64;        // 469
  const int mlpBlocksC = NC_  / 64;        // 1875
  const int gruBlocksV = (NVP_ / 32) * 2;  // 1876
  const int gruBlocksC = (NC_  / 32) * 2;  // 7500

  for (int it = 0; it < NITER; ++it) {
    // ---- v -> c ----
    hipMemsetAsync(aggr_c, 0, (size_t)NC_ * 2 * DD * 4, stream);
    mlp_fused_kernel<<<mlpBlocksV, 256, 0, stream>>>(vcur, mlpP[0][0], mlpW[0][1], mlpP[0][1], mlpW[0][3], msg_p);
    mlp_fused_kernel<<<mlpBlocksV, 256, 0, stream>>>(vcur, mlpP[1][0], mlpW[1][1], mlpP[1][1], mlpW[1][3], msg_n);
    scatter_kernel<<<scatterP, 256, 0, stream>>>(msg_p, pv, pc, aggr_c, 0, NEP_);
    scatter_kernel<<<scatterN, 256, 0, stream>>>(msg_n, nv, nc, aggr_c, DD, NEN_);
    gru_fused_kernel<<<gruBlocksC, 128, 0, stream>>>(
        aggr_c, pcinv, ncinv, ccur, gcWihP, gcWhhP, gc_bih, gc_bhh, cnext);

    // ---- c -> v ----
    hipMemsetAsync(aggr_v, 0, (size_t)NVP_ * 2 * DD * 4, stream);
    mlp_fused_kernel<<<mlpBlocksC, 256, 0, stream>>>(cnext, mlpP[2][0], mlpW[2][1], mlpP[2][1], mlpW[2][3], msg_p);
    mlp_fused_kernel<<<mlpBlocksC, 256, 0, stream>>>(cnext, mlpP[3][0], mlpW[3][1], mlpP[3][1], mlpW[3][3], msg_n);
    scatter_kernel<<<scatterP, 256, 0, stream>>>(msg_p, pc, pv, aggr_v, 0, NEP_);
    scatter_kernel<<<scatterN, 256, 0, stream>>>(msg_n, nc, nv, aggr_v, DD, NEN_);
    gru_fused_kernel<<<gruBlocksV, 128, 0, stream>>>(
        aggr_v, pvinv, nvinv, vcur, gvWihP, gvWhhP, gv_bih, gv_bhh, vnext);

    float* t;
    t = ccur; ccur = cnext; cnext = t;
    t = vcur; vcur = vnext; vnext = t;
  }

  // ---- readout (real rows only) ----
  hipMemsetAsync(gsum, 0, (size_t)NB_ * DD * 4, stream);
  hipMemsetAsync(gcnt, 0, (size_t)NB_ * 4, stream);
  readout_sum_kernel<<<(NV_ * 32 + 255) / 256, 256, 0, stream>>>(vcur, v_batch, gsum, gcnt, NV_);
  readout_mlp_kernel<<<1, 128, 0, stream>>>(gsum, gcnt, ro_W1, ro_b1, ro_W2, ro_b2, (float*)d_out);
}
